// PureTriXFFTLayer_63806034149900
// MI455X (gfx1250) — matrix operand order repaired and checked by reading, not compiled, not run on hardware
//
#include <hip/hip_runtime.h>
#include <hip/hip_bf16.h>
#include <math.h>

// Problem constants (match reference)
#define BATCH 131072
#define DMODEL 64
#define NTILE 4

typedef __bf16 bf16_t;
typedef __attribute__((ext_vector_type(16))) __bf16 v16bf;
typedef __attribute__((ext_vector_type(8)))  __bf16 v8bf;
typedef __attribute__((ext_vector_type(8)))  float  v8f;

// ---------------------------------------------------------------------------
// Padded bf16 weight layouts in d_ws (element counts). Row strides are odd
// multiples of 8 bf16 so 16-lane B-fragment reads from LDS hit distinct banks.
// ---------------------------------------------------------------------------
#define W1P_STRIDE 136               // 128 valid cols + 8 pad
#define W2P_STRIDE 72                // 64 valid + 8 pad
#define WR1P_STRIDE 72               // 64 valid + 8 pad
#define WR2P_STRIDE 24               // 16 valid (4 real + 12 zero) + 8 pad

#define W1P_TILE_ELEMS  (128 * W1P_STRIDE)          // 17408 per tile
#define W2P_TILE_ELEMS  (128 * W2P_STRIDE)          // 9216 per tile
#define W1P_ELEMS  (4 * W1P_TILE_ELEMS)             // 69632
#define W2P_ELEMS  (4 * W2P_TILE_ELEMS)             // 36864
#define WR1P_ELEMS (160 * WR1P_STRIDE)              // 11520 (K padded 136->160)
#define WR2P_ELEMS (64 * WR2P_STRIDE)               // 1536
#define WS_TOTAL_ELEMS (W1P_ELEMS + W2P_ELEMS + WR1P_ELEMS + WR2P_ELEMS) // 119552

// LDS plan (bytes): sW (max of router 26112 / tile 53248) + hbuf + lbuf + sIdx
#define SW_ELEMS   (W1P_TILE_ELEMS + W2P_TILE_ELEMS)   // 26624 bf16 = 53248 B
#define HBUF_ELEMS (8 * 16 * 136)                      // 17408 bf16 = 34816 B
#define SMEM_BYTES ((SW_ELEMS + HBUF_ELEMS) * 2 + 8 * 16 * 4 * 4 + 2 * 8 * 16 * 4)

// ---------------------------------------------------------------------------
// Kernel 0: convert f32 weights -> padded bf16 workspace
// ---------------------------------------------------------------------------
__global__ void prep_weights(const float* __restrict__ W1,
                             const float* __restrict__ W2,
                             const float* __restrict__ Wr1,
                             const float* __restrict__ Wr2,
                             bf16_t* __restrict__ ws) {
  int i = blockIdx.x * blockDim.x + threadIdx.x;
  if (i >= WS_TOTAL_ELEMS) return;
  float v = 0.0f;
  if (i < W1P_ELEMS) {
    int t = i / W1P_TILE_ELEMS, j = i % W1P_TILE_ELEMS;
    int r = j / W1P_STRIDE, c = j % W1P_STRIDE;
    if (c < 128) v = W1[(t * 128 + r) * 128 + c];
  } else if (i < W1P_ELEMS + W2P_ELEMS) {
    int j = i - W1P_ELEMS;
    int t = j / W2P_TILE_ELEMS; j %= W2P_TILE_ELEMS;
    int r = j / W2P_STRIDE, c = j % W2P_STRIDE;
    if (c < 64) v = W2[(t * 128 + r) * 64 + c];
  } else if (i < W1P_ELEMS + W2P_ELEMS + WR1P_ELEMS) {
    int j = i - (W1P_ELEMS + W2P_ELEMS);
    int r = j / WR1P_STRIDE, c = j % WR1P_STRIDE;
    if (r < 136 && c < 64) v = Wr1[r * 64 + c];
  } else {
    int j = i - (W1P_ELEMS + W2P_ELEMS + WR1P_ELEMS);
    int r = j / WR2P_STRIDE, c = j % WR2P_STRIDE;
    if (c < 4) v = Wr2[r * 4 + c];
  }
  ws[i] = (bf16_t)v;
}

// ---------------------------------------------------------------------------
// Helpers
// ---------------------------------------------------------------------------
__device__ __forceinline__ v8f wmma_bf16(v16bf a, v16bf b, v8f c) {
  return __builtin_amdgcn_wmma_f32_16x16x32_bf16(false, a, false, b,
                                                 (short)0, c, false, false);
}

// Branchless GELU (tanh form): x * sigmoid(1.59576912 x + 0.07135482 x^3).
// v_exp_f32/v_rcp_f32 on the TRANS pipe -> no exec-mask branches.
__device__ __forceinline__ float gelu_fast(float x) {
  float t = x * __builtin_fmaf(0.07135482f, x * x, 1.59576912f);
  float e = __builtin_amdgcn_exp2f(-1.44269504f * t);
  return x * __builtin_amdgcn_rcpf(1.0f + e);
}

// Combine two 16B halves into a 32B fragment register set.
__device__ __forceinline__ v16bf combine(v8bf lo, v8bf hi) {
  v16bf r;
#pragma unroll
  for (int e = 0; e < 8; ++e) { r[e] = lo[e]; r[8 + e] = hi[e]; }
  return r;
}

// B fragment (32x16): lane l holds column N = ntile*16 + (l&15); element e ->
// K = kstep*32 + (e/8)*16 + (l>>4)*8 + e%8. base is LDS, row-major [K][nstride] bf16.
__device__ __forceinline__ v16bf loadB_lds(const bf16_t* base, int nstride,
                                           int kstep, int ntile, int m, int half) {
  const bf16_t* p = base + (kstep * 32 + half * 8) * nstride + ntile * 16 + m;
  v16bf r;
#pragma unroll
  for (int e = 0; e < 16; ++e) r[e] = p[((e >> 3) * 16 + (e & 7)) * nstride];
  return r;
}

// A fragment (16x32) from LDS h buffer row (stride 136 bf16):
// lane holds row M = l&15; element e -> K = k2*32 + (e/8)*16 + half*8 + e%8.
__device__ __forceinline__ v16bf loadA_lds(const bf16_t* hrow, int k2, int half) {
  return combine(*(const v8bf*)(hrow + k2 * 32 + half * 8),
                 *(const v8bf*)(hrow + k2 * 32 + 16 + half * 8));
}

// Cooperative contiguous global->LDS copy (elems must be a multiple of 8 bf16).
__device__ __forceinline__ void stage_lds(bf16_t* dst, const bf16_t* src,
                                          int elems, int tid) {
  const uint4* s = (const uint4*)src;
  uint4* d = (uint4*)dst;
  int n = elems >> 3;                     // 8 bf16 per uint4
  for (int i = tid; i < n; i += 256) d[i] = s[i];
}

// ---------------------------------------------------------------------------
// Kernel 1: fused router + all-tile expert MLP + routed gather.
// One wave = one 16-row sample tile; 8 waves/block share LDS-staged weights.
// ---------------------------------------------------------------------------
__launch_bounds__(256)
__global__ void fused_router_moe(const float* __restrict__ a_vec,
                                 const float* __restrict__ b_vec,
                                 const float* __restrict__ pos,
                                 const float* __restrict__ br1,
                                 const float* __restrict__ br2,
                                 const float* __restrict__ b1,
                                 const float* __restrict__ b2,
                                 const bf16_t* __restrict__ ws,
                                 float* __restrict__ out) {
  extern __shared__ char smem[];
  bf16_t* sW   = (bf16_t*)smem;                 // staged weights (router / tile)
  bf16_t* hb   = sW + SW_ELEMS;                 // [8][16][136] D->A bounce
  float*  lbuf = (float*)(hb + HBUF_ELEMS);     // [8][16][4] router logits
  int*    sIdx = (int*)(lbuf + 8 * 16 * 4);     // [2][8][16] argmax ids

  const bf16_t* W1p  = ws;
  const bf16_t* W2p  = W1p + W1P_ELEMS;
  const bf16_t* Wr1p = W2p + W2P_ELEMS;
  const bf16_t* Wr2p = Wr1p + WR1P_ELEMS;

  const int tid  = threadIdx.x;
  const int w    = tid >> 5;
  const int lane = tid & 31;
  const int m    = lane & 15;
  const int half = lane >> 4;
  const long rowBase = ((long)blockIdx.x * 8 + w) * 16;

  const float* aRow = a_vec + (rowBase + m) * 64;
  const float* bRow = b_vec + (rowBase + m) * 64;
  const float* pRow = pos   + (rowBase + m) * 8;
  bf16_t* hrow = hb + (w * 16 + m) * 136;       // this lane's A-read row
  bf16_t* hwr  = hb + w * 16 * 136;             // this wave's h tile base

  // ---- Stage router weights while A-fragment loads are in flight ----
  stage_lds(sW, Wr1p, WR1P_ELEMS, tid);
  stage_lds(sW + WR1P_ELEMS, Wr2p, WR2P_ELEMS, tid);

  // ---- Build pair = [a|b] A-fragments (K = 0..127, 4 K-steps of 32) ----
  v16bf A[4];
#pragma unroll
  for (int f = 0; f < 4; ++f) {
#pragma unroll
    for (int g = 0; g < 2; ++g) {
      int kb = f * 32 + g * 16 + half * 8;      // multiple of 8, never spans a|b
      const float* src = (kb < 64) ? (aRow + kb) : (bRow + kb - 64);
      float4 lo = *(const float4*)src;
      float4 hi = *(const float4*)(src + 4);
      A[f][g * 8 + 0] = (bf16_t)lo.x; A[f][g * 8 + 1] = (bf16_t)lo.y;
      A[f][g * 8 + 2] = (bf16_t)lo.z; A[f][g * 8 + 3] = (bf16_t)lo.w;
      A[f][g * 8 + 4] = (bf16_t)hi.x; A[f][g * 8 + 5] = (bf16_t)hi.y;
      A[f][g * 8 + 6] = (bf16_t)hi.z; A[f][g * 8 + 7] = (bf16_t)hi.w;
    }
  }
  // pos A-fragment: K = 128..159; only K=128..135 valid (lanes 0-15, e<8)
  v16bf P;
#pragma unroll
  for (int e = 0; e < 16; ++e) P[e] = (bf16_t)0.0f;
  if (half == 0) {
    float4 p0 = *(const float4*)pRow;
    float4 p1 = *(const float4*)(pRow + 4);
    P[0] = (bf16_t)p0.x; P[1] = (bf16_t)p0.y; P[2] = (bf16_t)p0.z; P[3] = (bf16_t)p0.w;
    P[4] = (bf16_t)p1.x; P[5] = (bf16_t)p1.y; P[6] = (bf16_t)p1.z; P[7] = (bf16_t)p1.w;
  }

  __syncthreads();                              // router weights visible

  const bf16_t* sWr1 = sW;
  const bf16_t* sWr2 = sW + WR1P_ELEMS;

  // ---- Routers: rt=0 uses [a|b|pos], rt=1 uses [b|a|pos] (frags reordered) --
#pragma unroll
  for (int rt = 0; rt < 2; ++rt) {
#pragma unroll
    for (int nt = 0; nt < 4; ++nt) {            // h = gelu(x @ Wr1 + br1)
      v8f acc = {};
#pragma unroll
      for (int k = 0; k < 5; ++k) {
        v16bf af = (k == 4) ? P : A[(k + 2 * rt) & 3];
        acc = wmma_bf16(af, loadB_lds(sWr1, WR1P_STRIDE, k, nt, m, half), acc);
      }
      float bb = br1[nt * 16 + m];
#pragma unroll
      for (int r = 0; r < 8; ++r) {
        int M = r + half * 8;                   // D layout: lanes16-31 -> M+8
        hwr[M * 136 + nt * 16 + m] = (bf16_t)gelu_fast(acc[r] + bb);
      }
    }
    v8f lacc = {};                              // logits = h @ Wr2 (+ br2)
#pragma unroll
    for (int k2 = 0; k2 < 2; ++k2) {
      v16bf a2 = loadA_lds(hrow, k2, half);
      lacc = wmma_bf16(a2, loadB_lds(sWr2, WR2P_STRIDE, k2, 0, m, half), lacc);
    }
#pragma unroll
    for (int r = 0; r < 8; ++r) {
      int M = r + half * 8;
      if (m < 4) lbuf[(w * 16 + M) * 4 + m] = lacc[r] + br2[m];
    }
    __builtin_amdgcn_wave_barrier();
    if (lane < 16) {                            // argmax over 4 (first-max tie)
      const float* lr = &lbuf[(w * 16 + lane) * 4];
      float best = lr[0];
      int bi = 0;
#pragma unroll
      for (int n = 1; n < 4; ++n) {
        float v = lr[n];
        if (v > best) { best = v; bi = n; }
      }
      sIdx[(rt * 8 + w) * 16 + lane] = bi;
    }
    __builtin_amdgcn_wave_barrier();
  }

  int id1[8], id2[8];
#pragma unroll
  for (int r = 0; r < 8; ++r) {
    int M = r + half * 8;
    id1[r] = sIdx[(0 * 8 + w) * 16 + M];
    id2[r] = sIdx[(1 * 8 + w) * 16 + M];
  }

  float* out1 = out;
  float* out2 = out + (long)BATCH * DMODEL;
  bf16_t* sW2 = sW + W1P_TILE_ELEMS;

  // ---- Expert tiles: stage W1t+W2t per tile, run all T, scatter routed rows -
  for (int t = 0; t < NTILE; ++t) {
    __syncthreads();                            // previous phase's reads done
    stage_lds(sW, W1p + t * W1P_TILE_ELEMS, W1P_TILE_ELEMS, tid);
    stage_lds(sW2, W2p + t * W2P_TILE_ELEMS, W2P_TILE_ELEMS, tid);
    __syncthreads();                            // tile weights visible

#pragma unroll
    for (int nt = 0; nt < 8; ++nt) {            // h = gelu(pair @ W1[t] + b1)
      v8f acc = {};
#pragma unroll
      for (int k = 0; k < 4; ++k)
        acc = wmma_bf16(A[k], loadB_lds(sW, W1P_STRIDE, k, nt, m, half), acc);
      float bb = b1[t * 128 + nt * 16 + m];
#pragma unroll
      for (int r = 0; r < 8; ++r) {
        int M = r + half * 8;
        hwr[M * 136 + nt * 16 + m] = (bf16_t)gelu_fast(acc[r] + bb);
      }
    }
#pragma unroll
    for (int nt = 0; nt < 4; ++nt) {            // out_t = h @ W2[t] + b2
      v8f acc = {};
#pragma unroll
      for (int k2 = 0; k2 < 4; ++k2) {
        v16bf a2 = loadA_lds(hrow, k2, half);
        acc = wmma_bf16(a2, loadB_lds(sW2, W2P_STRIDE, k2, nt, m, half), acc);
      }
      float bb = b2[t * 64 + nt * 16 + m];
#pragma unroll
      for (int r = 0; r < 8; ++r) {
        int M = r + half * 8;
        float o = acc[r] + bb;
        long off = (rowBase + M) * 64 + nt * 16 + m;
        if (id1[r] == t) out1[off] = o;
        if (id2[r] == t) out2[off] = o;
      }
    }
  }
}

// ---------------------------------------------------------------------------
// Host launcher
// ---------------------------------------------------------------------------
extern "C" void kernel_launch(void* const* d_in, const int* in_sizes, int n_in,
                              void* d_out, int out_size, void* d_ws, size_t ws_size,
                              hipStream_t stream) {
  (void)in_sizes; (void)n_in; (void)out_size; (void)ws_size;
  const float* a_vec = (const float*)d_in[0];
  const float* b_vec = (const float*)d_in[1];
  const float* posE  = (const float*)d_in[2];
  const float* Wr1   = (const float*)d_in[3];
  const float* br1   = (const float*)d_in[4];
  const float* Wr2   = (const float*)d_in[5];
  const float* br2   = (const float*)d_in[6];
  const float* W1    = (const float*)d_in[7];
  const float* b1    = (const float*)d_in[8];
  const float* W2    = (const float*)d_in[9];
  const float* b2    = (const float*)d_in[10];
  bf16_t* ws  = (bf16_t*)d_ws;
  float*  out = (float*)d_out;

  // Allow ~91KB dynamic LDS (idempotent; safe under graph capture).
  hipFuncSetAttribute((const void*)fused_router_moe,
                      hipFuncAttributeMaxDynamicSharedMemorySize, SMEM_BYTES);

  prep_weights<<<(WS_TOTAL_ELEMS + 255) / 256, 256, 0, stream>>>(W1, W2, Wr1, Wr2, ws);
  fused_router_moe<<<BATCH / 128, 256, SMEM_BYTES, stream>>>(
      a_vec, b_vec, posE, br1, br2, b1, b2, ws, out);
}
